// SDCN_704374636673
// MI455X (gfx1250) — compile-verified, matrix operand order checked by reading, not run on hardware
//
#include <hip/hip_runtime.h>
#include <stdint.h>

// ---------------- problem constants (match reference) ----------------
#define N_NODES 100000
#define D_IN    500
#define DE1     500
#define DE2     500
#define DE3     2000
#define NZ      10
#define KCL     10
#define NEDGE   3200000

typedef __attribute__((ext_vector_type(16))) __bf16 v16bf;
typedef __attribute__((ext_vector_type(8)))  float  v8f;

union BFrag { v16bf v; uint32_t u[8]; };
union CFrag { v8f   v; float    f[8]; };

// f32 -> bf16 (RNE), as raw u16
__device__ inline uint16_t rne16(float f) {
  uint32_t u = __float_as_uint(f);
  return (uint16_t)((u + 0x7FFFu + ((u >> 16) & 1u)) >> 16);
}
// pack two f32 -> dword of 2 bf16 (lo=a, hi=b)
__device__ inline uint32_t pk2bf(float a, float b) {
  uint32_t ua = __float_as_uint(a), ub = __float_as_uint(b);
  ua = (ua + 0x7FFFu + ((ua >> 16) & 1u)) >> 16;
  ub = (ub + 0x7FFFu + ((ub >> 16) & 1u)) & 0xFFFF0000u;
  return ua | ub;
}
__device__ inline float bflo(uint32_t u) { return __uint_as_float(u << 16); }
__device__ inline float bfhi(uint32_t u) { return __uint_as_float(u & 0xFFFF0000u); }
__device__ inline float u16f(uint16_t u) { return __uint_as_float((uint32_t)u << 16); }

// LDS byte offset of a __shared__ object = low 32 bits of its generic address
__device__ inline uint32_t lds_off(const void* p) { return (uint32_t)(uintptr_t)p; }

// ---------------- WMMA GEMM: C = act( A @ Wt^T + bias ) ----------------
// A: [nrows x K] bf16. Wt: [M x K] bf16 (pre-transposed weights).
// Block tile 128x128, 8 waves (wave32), wave = 32x64 = 2x4 v_wmma_f32_16x16x32_bf16.
// Interior tiles stream global->LDS via GLOBAL_LOAD_ASYNC_TO_LDS_B64 into a
// double-buffered LDS tile (ASYNCcnt + barrier sync) so tile k+1 copies while
// tile k feeds the matrix pipe. Boundary tiles use a guarded register path.
#define BM 128
#define BN 128
#define BK 32
#define LDK 20   // padded row stride in dwords (= 40 bf16)

__global__ __launch_bounds__(256)
void gemm_bf16(const uint16_t* __restrict__ A, const uint16_t* __restrict__ Wt,
               const float* __restrict__ bias, void* __restrict__ Cout,
               int nrows, int K, int M, int doRelu, int outBF)
{
  __shared__ uint32_t As[2][BM][LDK];
  __shared__ uint32_t Bs[2][BN][LDK];

  const int tid  = threadIdx.x;
  const int wave = tid >> 5;
  const int lane = tid & 31;
  const int lm   = lane & 15;
  const int lh   = lane >> 4;
  const int wr   = wave >> 1;        // 0..3 : 32-row strip
  const int wc   = wave & 1;         // 0..1 : 64-col strip
  const int r0   = blockIdx.y * BM;
  const int c0   = blockIdx.x * BN;

  // per-thread staging map: 4 chunks of 8B (4 bf16); rl = row, cu = k-chunk
  auto stage = [&](const uint16_t* __restrict__ P, uint32_t (*S)[LDK],
                   int rows, int rbase, int k0) {
    const bool fast = (rbase + BM <= rows) && (k0 + BK <= K) && ((K & 3) == 0);
    if (fast) {
      #pragma unroll
      for (int p = 0; p < 4; ++p) {
        int id = tid + p * 256;
        int rl = id >> 3, cu = id & 7;
        uint32_t lds = lds_off(&S[rl][cu * 2]);
        uint64_t ga  = (uint64_t)(uintptr_t)(P + (size_t)(rbase + rl) * K + k0 + cu * 4);
        asm volatile("global_load_async_to_lds_b64 %0, %1, off"
                     :: "v"(lds), "v"(ga) : "memory");
      }
    } else {
      #pragma unroll
      for (int p = 0; p < 4; ++p) {
        int id = tid + p * 256;
        int rl = id >> 3, cu = id & 7;
        int gr = rbase + rl;
        uint32_t w[4];
        #pragma unroll
        for (int t = 0; t < 4; ++t) {
          int gk = k0 + cu * 4 + t;
          w[t] = (gr < rows && gk < K) ? (uint32_t)P[(size_t)gr * K + gk] : 0u;
        }
        *(uint2*)&S[rl][cu * 2] = make_uint2(w[0] | (w[1] << 16), w[2] | (w[3] << 16));
      }
    }
  };

  CFrag c[2][4];
  #pragma unroll
  for (int i = 0; i < 2; ++i)
    #pragma unroll
    for (int j = 0; j < 4; ++j)
      #pragma unroll
      for (int r = 0; r < 8; ++r) c[i][j].f[r] = 0.f;

  const int nk = (K + BK - 1) / BK;
  stage(A,  As[0], nrows, r0, 0);
  stage(Wt, Bs[0], M,     c0, 0);
  for (int kt = 0; kt < nk; ++kt) {
    asm volatile("s_wait_asynccnt 0x0" ::: "memory");   // async copies of this tile done
    __syncthreads();                                    // visible to all waves
    const int cur = kt & 1;
    if (kt + 1 < nk) {                                  // stream next tile into other buffer
      stage(A,  As[cur ^ 1], nrows, r0, (kt + 1) * BK);
      stage(Wt, Bs[cur ^ 1], M,     c0, (kt + 1) * BK);
    }

    BFrag a[2], b[4];
    #pragma unroll
    for (int i = 0; i < 2; ++i) {
      const uint32_t* pr = &As[cur][wr * 32 + i * 16 + lm][0];
      *(uint4*)&a[i].u[0] = *(const uint4*)&pr[lh * 4];
      *(uint4*)&a[i].u[4] = *(const uint4*)&pr[8 + lh * 4];
    }
    #pragma unroll
    for (int j = 0; j < 4; ++j) {
      const uint32_t* pr = &Bs[cur][wc * 64 + j * 16 + lm][0];
      *(uint4*)&b[j].u[0] = *(const uint4*)&pr[lh * 4];
      *(uint4*)&b[j].u[4] = *(const uint4*)&pr[8 + lh * 4];
    }
    #pragma unroll
    for (int i = 0; i < 2; ++i)
      #pragma unroll
      for (int j = 0; j < 4; ++j)
        c[i][j].v = __builtin_amdgcn_wmma_f32_16x16x32_bf16(
            false, a[i].v, false, b[j].v, (short)0, c[i][j].v, false, false);
  }

  // epilogue (single conversion per output element when outBF)
  #pragma unroll
  for (int j = 0; j < 4; ++j) {
    int colg = c0 + wc * 64 + j * 16 + lm;
    if (colg < M) {
      float bv = bias ? bias[colg] : 0.f;
      #pragma unroll
      for (int i = 0; i < 2; ++i) {
        #pragma unroll
        for (int r = 0; r < 8; ++r) {
          int rowg = r0 + wr * 32 + i * 16 + lh * 8 + r;
          if (rowg < nrows) {
            float v = c[i][j].f[r] + bv;
            if (doRelu) v = fmaxf(v, 0.f);
            if (outBF) ((uint16_t*)Cout)[(size_t)rowg * M + colg] = rne16(v);
            else       ((float*)Cout)[(size_t)rowg * M + colg] = v;
          }
        }
      }
    }
  }
}

// ---------------- weight transpose + convert: Wt[m][k] = bf16(W[k][m]) ----------------
__global__ void conv_w_t(const float* __restrict__ in, uint16_t* __restrict__ out, int K, int M) {
  int i = blockIdx.x * 256 + threadIdx.x;
  if (i < K * M) {
    int k = i / M, m = i % M;
    out[(size_t)m * K + k] = rne16(in[(size_t)k * M + m]);
  }
}

// f32 -> bf16 bulk convert (n % 4 == 0)
__global__ void cvt_bf16(const float4* __restrict__ in, uint2* __restrict__ out, long n4) {
  long i = (long)blockIdx.x * 256 + threadIdx.x;
  if (i < n4) {
    float4 v = in[i];
    out[i] = make_uint2(pk2bf(v.x, v.y), pk2bf(v.z, v.w));
  }
}

// combine: out_bf = al*a(f32) + be*b(bf16)
__global__ void combine_fb(const float4* __restrict__ a, const uint2* __restrict__ b,
                           uint2* __restrict__ o, long n4, float al, float be) {
  long i = (long)blockIdx.x * 256 + threadIdx.x;
  if (i < n4) {
    float4 av = a[i]; uint2 bv = b[i];
    float c0 = al * av.x + be * bflo(bv.x);
    float c1 = al * av.y + be * bfhi(bv.x);
    float c2 = al * av.z + be * bflo(bv.y);
    float c3 = al * av.w + be * bfhi(bv.y);
    o[i] = make_uint2(pk2bf(c0, c1), pk2bf(c2, c3));
  }
}

// ---------------- SpMM: out[r] += val * m[c] ; m is bf16 (halved gather bytes) ----------------
__global__ void spmm_row(const int* __restrict__ rows, const int* __restrict__ cols,
                         const float* __restrict__ vals, const uint16_t* __restrict__ m,
                         float* __restrict__ out, int W)
{
  int e = blockIdx.x;                   // uniform -> scalarized
  int r = rows[e], c = cols[e];
  float v = vals[e];
  const uint2* src = (const uint2*)(m + (size_t)c * W);
  float* dst = out + (size_t)r * W;
  int W4 = W >> 2;
  for (int j = threadIdx.x; j < W4; j += blockDim.x) {
    uint2 s = src[j];
    atomicAdd(&dst[j*4+0], v * bflo(s.x));
    atomicAdd(&dst[j*4+1], v * bfhi(s.x));
    atomicAdd(&dst[j*4+2], v * bflo(s.y));
    atomicAdd(&dst[j*4+3], v * bfhi(s.y));
  }
}

__global__ void spmm_small(const int* __restrict__ rows, const int* __restrict__ cols,
                           const float* __restrict__ vals, const uint16_t* __restrict__ m,
                           float* __restrict__ out, int W)
{
  long gid = (long)blockIdx.x * blockDim.x + threadIdx.x;
  int e = (int)(gid >> 4);
  int j = (int)(gid & 15);
  if (e < NEDGE && j < W) {
    int r = rows[e], c = cols[e];
    atomicAdd(&out[(size_t)r * W + j], vals[e] * u16f(m[(size_t)c * W + j]));
  }
}

__global__ void relu4_k(float4* p, long n4) {
  long i = (long)blockIdx.x * blockDim.x + threadIdx.x;
  if (i < n4) {
    float4 v = p[i];
    v.x = fmaxf(v.x, 0.f); v.y = fmaxf(v.y, 0.f);
    v.z = fmaxf(v.z, 0.f); v.w = fmaxf(v.w, 0.f);
    p[i] = v;
  }
}

__global__ void softmax10(const float* __restrict__ in, float* __restrict__ out) {
  int r = blockIdx.x * blockDim.x + threadIdx.x;
  if (r >= N_NODES) return;
  float v[KCL]; float mx = -3.4e38f;
  #pragma unroll
  for (int k = 0; k < KCL; ++k) { v[k] = in[r * KCL + k]; mx = fmaxf(mx, v[k]); }
  float s = 0.f;
  #pragma unroll
  for (int k = 0; k < KCL; ++k) { v[k] = __expf(v[k] - mx); s += v[k]; }
  float inv = 1.f / s;
  #pragma unroll
  for (int k = 0; k < KCL; ++k) out[r * KCL + k] = v[k] * inv;
}

// Student-t, V=1: q = 1/(1+d), normalized
__global__ void student_q(const float* __restrict__ z, const float* __restrict__ cluster,
                          float* __restrict__ q)
{
  int r = blockIdx.x * blockDim.x + threadIdx.x;
  if (r >= N_NODES) return;
  float zr[NZ];
  #pragma unroll
  for (int j = 0; j < NZ; ++j) zr[j] = z[r * NZ + j];
  float qv[KCL]; float s = 0.f;
  #pragma unroll
  for (int k = 0; k < KCL; ++k) {
    float d = 0.f;
    #pragma unroll
    for (int j = 0; j < NZ; ++j) { float t = zr[j] - cluster[k * NZ + j]; d += t * t; }
    float qq = 1.f / (1.f + d);
    qv[k] = qq; s += qq;
  }
  float inv = 1.f / s;
  #pragma unroll
  for (int k = 0; k < KCL; ++k) q[r * KCL + k] = qv[k] * inv;
}

// ---------------- launcher ----------------
extern "C" void kernel_launch(void* const* d_in, const int* in_sizes, int n_in,
                              void* d_out, int out_size, void* d_ws, size_t ws_size,
                              hipStream_t stream)
{
  const float* x      = (const float*)d_in[0];
  const int*   arows  = (const int*)  d_in[1];
  const int*   acols  = (const int*)  d_in[2];
  const float* avals  = (const float*)d_in[3];
  const float* wsrc[13] = {
    (const float*)d_in[4],  (const float*)d_in[6],  (const float*)d_in[8],  (const float*)d_in[10],
    (const float*)d_in[12], (const float*)d_in[14], (const float*)d_in[16], (const float*)d_in[18],
    (const float*)d_in[20], (const float*)d_in[21], (const float*)d_in[22], (const float*)d_in[23],
    (const float*)d_in[24] };
  const int wK[13] = { D_IN, DE1, DE2, DE3, NZ, DE3, DE2, DE1, D_IN, DE1, DE2, DE3, NZ };
  const int wM[13] = { DE1,  DE2, DE3, NZ,  DE3, DE2, DE1, D_IN, DE1, DE2, DE3, NZ,  KCL };
  const float* enc1_b = (const float*)d_in[5];
  const float* enc2_b = (const float*)d_in[7];
  const float* enc3_b = (const float*)d_in[9];
  const float* z_b    = (const float*)d_in[11];
  const float* dec1_b = (const float*)d_in[13];
  const float* dec2_b = (const float*)d_in[15];
  const float* dec3_b = (const float*)d_in[17];
  const float* xbar_b = (const float*)d_in[19];
  const float* clus   = (const float*)d_in[25];

  float* out  = (float*)d_out;
  float* xbar = out;                                        // [N x 500]
  float* qout = out + (size_t)N_NODES * D_IN;               // [N x 10]
  float* pred = qout + (size_t)N_NODES * NZ;                // [N x 10]
  float* zbuf = pred + (size_t)N_NODES * NZ;                // [N x 10] (f32)

  // ---- workspace carve (bf16 stored as u16); all region sizes multiples of 8 elems ----
  uint16_t* wsu = (uint16_t*)d_ws;
  uint16_t* wt[13];
  {
    size_t off = 0;
    for (int i = 0; i < 13; ++i) { wt[i] = wsu + off; off += ((size_t)wK[i] * wM[i] + 7) & ~(size_t)7; }
  }
  uint16_t* xbf  = wsu + 5000000;            // N x 500
  uint16_t* h1   = xbf  +  50000000ll;       // N x 500
  uint16_t* h2   = h1   +  50000000ll;       // N x 500
  uint16_t* h3   = h2   +  50000000ll;       // N x 2000
  uint16_t* zbf  = h3   + 200000000ll;       // N x 10
  uint16_t* bufT = zbf  +   1000000ll;       // N x 2000 (pre-spmm GEMM outs; decoder d1)
  uint16_t* cmb  = bufT + 200000000ll;       // N x 2000 (combine outs; decoder d2/d3)
  uint16_t* t4b  = cmb  + 200000000ll;       // N x 10
  uint16_t* t5b  = t4b  +   1000000ll;       // N x 10
  uint16_t* c5b  = t5b  +   1000000ll;       // N x 10
  float*    bufH = (float*)(c5b + 1000000ll);// N x 2000 f32 (spmm outputs hg1/hg2/hg3)
  float*    hg4f = bufH + 200000000ll;       // N x 10 f32
  float*    hpre = hg4f +   1000000ll;       // N x 10 f32

  const dim3 blk(256);
  auto grid = [](int M) { return dim3((unsigned)((M + BN - 1) / BN), (unsigned)((N_NODES + BM - 1) / BM)); };
  const float sg = 0.5f;   // SIGMA
  const unsigned rowsK = (unsigned)((N_NODES + 255) / 256);
  auto relu4 = [&](float* p, long n) {
    relu4_k<<<(unsigned)((n / 4 + 255) / 256), blk, 0, stream>>>((float4*)p, n / 4);
  };
  auto comb = [&](const float* a, const uint16_t* b, uint16_t* o, long n) {
    combine_fb<<<(unsigned)((n / 4 + 255) / 256), blk, 0, stream>>>(
        (const float4*)a, (const uint2*)b, (uint2*)o, n / 4, 1.f - sg, sg);
  };

  // ---- one-time converts: weights (transpose+bf16), x -> bf16 ----
  for (int i = 0; i < 13; ++i)
    conv_w_t<<<(unsigned)(((long)wK[i] * wM[i] + 255) / 256), blk, 0, stream>>>(wsrc[i], wt[i], wK[i], wM[i]);
  cvt_bf16<<<(unsigned)(((long)N_NODES * D_IN / 4 + 255) / 256), blk, 0, stream>>>(
      (const float4*)x, (uint2*)xbf, (long)N_NODES * D_IN / 4);

  // ---- AE encoder ----
  gemm_bf16<<<grid(DE1), blk, 0, stream>>>(xbf, wt[0], enc1_b, h1, N_NODES, D_IN, DE1, 1, 1);
  gemm_bf16<<<grid(DE2), blk, 0, stream>>>(h1,  wt[1], enc2_b, h2, N_NODES, DE1, DE2, 1, 1);
  gemm_bf16<<<grid(DE3), blk, 0, stream>>>(h2,  wt[2], enc3_b, h3, N_NODES, DE2, DE3, 1, 1);
  gemm_bf16<<<grid(NZ),  blk, 0, stream>>>(h3,  wt[3], z_b,  zbuf, N_NODES, DE3, NZ,  0, 0);
  cvt_bf16<<<(unsigned)(((long)N_NODES * NZ / 4 + 255) / 256), blk, 0, stream>>>(
      (const float4*)zbuf, (uint2*)zbf, (long)N_NODES * NZ / 4);

  // ---- GNN ----
  // layer 1
  gemm_bf16<<<grid(DE1), blk, 0, stream>>>(xbf, wt[8], nullptr, bufT, N_NODES, D_IN, DE1, 0, 1);
  hipMemsetAsync(bufH, 0, (size_t)N_NODES * DE1 * sizeof(float), stream);
  spmm_row<<<NEDGE, blk, 0, stream>>>(arows, acols, avals, bufT, bufH, DE1);
  relu4(bufH, (long)N_NODES * DE1);
  // layer 2
  comb(bufH, h1, cmb, (long)N_NODES * DE1);
  gemm_bf16<<<grid(DE2), blk, 0, stream>>>(cmb, wt[9], nullptr, bufT, N_NODES, DE1, DE2, 0, 1);
  hipMemsetAsync(bufH + 50000000ll, 0, (size_t)N_NODES * DE2 * sizeof(float), stream);
  spmm_row<<<NEDGE, blk, 0, stream>>>(arows, acols, avals, bufT, bufH + 50000000ll, DE2);
  relu4(bufH + 50000000ll, (long)N_NODES * DE2);
  // layer 3
  comb(bufH + 50000000ll, h2, cmb, (long)N_NODES * DE2);
  gemm_bf16<<<grid(DE3), blk, 0, stream>>>(cmb, wt[10], nullptr, bufT, N_NODES, DE2, DE3, 0, 1);
  hipMemsetAsync(bufH, 0, (size_t)N_NODES * DE3 * sizeof(float), stream);
  spmm_row<<<NEDGE, blk, 0, stream>>>(arows, acols, avals, bufT, bufH, DE3);
  relu4(bufH, (long)N_NODES * DE3);
  // layer 4
  comb(bufH, h3, cmb, (long)N_NODES * DE3);
  gemm_bf16<<<grid(NZ), blk, 0, stream>>>(cmb, wt[11], nullptr, t4b, N_NODES, DE3, NZ, 0, 1);
  hipMemsetAsync(hg4f, 0, (size_t)N_NODES * NZ * sizeof(float), stream);
  spmm_small<<<(unsigned)(((long)NEDGE * 16 + 255) / 256), blk, 0, stream>>>(arows, acols, avals, t4b, hg4f, NZ);
  relu4(hg4f, (long)N_NODES * NZ);
  // layer 5 (no relu)
  comb(hg4f, zbf, c5b, (long)N_NODES * NZ);
  gemm_bf16<<<grid(KCL), blk, 0, stream>>>(c5b, wt[12], nullptr, t5b, N_NODES, NZ, KCL, 0, 1);
  hipMemsetAsync(hpre, 0, (size_t)N_NODES * KCL * sizeof(float), stream);
  spmm_small<<<(unsigned)(((long)NEDGE * 16 + 255) / 256), blk, 0, stream>>>(arows, acols, avals, t5b, hpre, KCL);
  softmax10<<<rowsK, blk, 0, stream>>>(hpre, pred);

  // ---- AE decoder (bufT/cmb free now) ----
  uint16_t* d2 = cmb;
  uint16_t* d3 = cmb + 50000000ll;
  gemm_bf16<<<grid(DE3),  blk, 0, stream>>>(zbf,  wt[4], dec1_b, bufT, N_NODES, NZ,  DE3, 1, 1);
  gemm_bf16<<<grid(DE2),  blk, 0, stream>>>(bufT, wt[5], dec2_b, d2,   N_NODES, DE3, DE2, 1, 1);
  gemm_bf16<<<grid(DE1),  blk, 0, stream>>>(d2,   wt[6], dec3_b, d3,   N_NODES, DE2, DE1, 1, 1);
  gemm_bf16<<<grid(D_IN), blk, 0, stream>>>(d3,   wt[7], xbar_b, xbar, N_NODES, DE1, D_IN, 0, 0);

  // ---- student-t q ----
  student_q<<<rowsK, blk, 0, stream>>>(zbuf, clus, qout);
}
